// AttentionHead_72069551227456
// MI455X (gfx1250) — matrix operand order repaired and checked by reading, compiled, not run on hardware
//
#include <hip/hip_runtime.h>
#include <hip/hip_bf16.h>
#include <math.h>

typedef __attribute__((ext_vector_type(16))) _Float16 v16h;
typedef __attribute__((ext_vector_type(8)))  float    v8f;

#define NN      10000
#define IN_F    512
#define OUT_F   128
#define NE      320000
#define WPR     313           // ceil(10000/32) mask words per row
#define NEG_SLOPE 0.2f

// ---------------- workspace layout (bytes) ----------------
#define OFF_MASK   0                         // 10000*313*4 = 12,520,000
#define OFF_H      12520192                  // 10000*128*4 = 5,120,000
#define OFF_S1     17640192                  // 40,000
#define OFF_S2     17680192                  // 40,000
#define OFF_HMEAN  17720192                  // 512
#define OFF_WF16   17720704                  // 128*512*2 = 131,072 (fragment-swizzled)

#define WF16_BYTES 131072

// ---------------- 1) zero the adjacency bitmask ----------------
__global__ void zero_mask_kernel(unsigned* __restrict__ mask, int nwords) {
    int t = blockIdx.x * blockDim.x + threadIdx.x;
    if (t < nwords) mask[t] = 0u;
}

// ---------------- 2) scatter edges into bitmask (dedups duplicates) ----------------
__global__ void build_mask_kernel(const long long* __restrict__ ei, unsigned* __restrict__ mask) {
    int e = blockIdx.x * blockDim.x + threadIdx.x;
    if (e >= NE) return;
    int i = (int)ei[e];        // source row
    int j = (int)ei[NE + e];   // dest col
    atomicOr(&mask[(size_t)i * WPR + (j >> 5)], 1u << (j & 31));
}

// ---------------- 3a) convert W -> f16, swizzled into B-fragment layout ----------------
// Fragment f = kb*8 + nt. For lane L: element e = W[nt*16+(L&15)][kb*32+(L>>4)*8+e+(e>>3)*8],
// stored at Wf16[(f*32 + L)*16 + e] -> each lane's fragment is one contiguous 32B chunk.
__global__ void convert_w_kernel(const float* __restrict__ W, _Float16* __restrict__ Wf16) {
    int t = blockIdx.x * blockDim.x + threadIdx.x;     // 4096 threads
    if (t >= 4096) return;
    int f    = t >> 5;
    int lane = t & 31;
    int kb   = f >> 3;
    int nt   = f & 7;
    int l16  = lane & 15;
    int hf   = lane >> 4;
    const float* src = W + (size_t)(nt * 16 + l16) * IN_F + kb * 32 + hf * 8;
    v16h b;
    #pragma unroll
    for (int q = 0; q < 4; ++q) {
        float4 wv = *(const float4*)(src + 4 * q + (q >> 1) * 8);
        b[4 * q]     = (_Float16)wv.x;
        b[4 * q + 1] = (_Float16)wv.y;
        b[4 * q + 2] = (_Float16)wv.z;
        b[4 * q + 3] = (_Float16)wv.w;
    }
    *(v16h*)(Wf16 + (size_t)t * 16) = b;
}

// A-fragment loader: two contiguous 32B chunks at k = hf*8 and k = 16+hf*8.
// Four independent float4 loads -> compiler can clause them and wait once.
__device__ __forceinline__ v16h load_a_frag(const float* p) {
    float4 f0 = *(const float4*)(p);
    float4 f1 = *(const float4*)(p + 4);
    float4 f2 = *(const float4*)(p + 16);
    float4 f3 = *(const float4*)(p + 20);
    v16h a;
    a[0]  = (_Float16)f0.x;  a[1]  = (_Float16)f0.y;
    a[2]  = (_Float16)f0.z;  a[3]  = (_Float16)f0.w;
    a[4]  = (_Float16)f1.x;  a[5]  = (_Float16)f1.y;
    a[6]  = (_Float16)f1.z;  a[7]  = (_Float16)f1.w;
    a[8]  = (_Float16)f2.x;  a[9]  = (_Float16)f2.y;
    a[10] = (_Float16)f2.z;  a[11] = (_Float16)f2.w;
    a[12] = (_Float16)f3.x;  a[13] = (_Float16)f3.y;
    a[14] = (_Float16)f3.z;  a[15] = (_Float16)f3.w;
    return a;
}

// ---------------- 3b) h = x @ W^T via WMMA f16 (fp32 accum) ----------------
// 4 waves/block, one 16-row M-tile per wave, all 8 N-tiles per wave (A reuse x8).
// Full f16 W (128KB) staged into dynamic LDS via gfx1250 async-to-LDS DMA.
__global__ void gemm_wmma_kernel(const float* __restrict__ x,
                                 const _Float16* __restrict__ Wf16,
                                 float* __restrict__ h) {
    extern __shared__ _Float16 sW[];                 // 65536 halves = 128KB

    // ---- stage W into LDS with async copies (ASYNCcnt-tracked) ----
    {
        unsigned ldsbase = (unsigned)(uintptr_t)sW;  // low 32 bits = LDS byte offset
        const char* gsrc = (const char*)Wf16;
        #pragma unroll 4
        for (int it = 0; it < WF16_BYTES / (256 * 16); ++it) {   // 32 iters of 16B/thread
            unsigned boff = (unsigned)((it * 256 + threadIdx.x) * 16);
            unsigned laddr = ldsbase + boff;
            unsigned long long gaddr = (unsigned long long)(uintptr_t)(gsrc + boff);
            asm volatile("global_load_async_to_lds_b128 %0, %1, off"
                         :: "v"(laddr), "v"(gaddr) : "memory");
        }
        asm volatile("s_wait_asynccnt 0" ::: "memory");
    }
    __syncthreads();

    const int wave  = threadIdx.x >> 5;
    const int lane  = threadIdx.x & 31;
    const int mtile = blockIdx.x * 4 + wave;         // 4 waves / block
    if (mtile >= NN / 16) return;                    // wave-uniform guard
    const int m0  = mtile * 16;
    const int l16 = lane & 15;
    const int hf  = lane >> 4;

    v8f acc[8];
    #pragma unroll
    for (int t = 0; t < 8; ++t) { v8f z{}; acc[t] = z; }

    const float* xr = x + (size_t)(m0 + l16) * IN_F + hf * 8;   // lane's A base

    for (int kb = 0; kb < IN_F / 32; ++kb) {
        const int k0 = kb * 32;
        if (k0 + 32 < IN_F) __builtin_prefetch(xr + k0 + 32, 0, 3);  // WGP-scope prefetch

        v16h a = load_a_frag(xr + k0);

        // 8 column tiles share this A fragment; B fragments direct from LDS
        #pragma unroll
        for (int nt = 0; nt < 8; ++nt) {
            const _Float16* bp = sW + ((size_t)((kb << 3) + nt) * 32 + lane) * 16;
            v16h b = *(const v16h*)bp;               // 2x ds_load_b128, pre-converted
            acc[nt] = __builtin_amdgcn_wmma_f32_16x16x32_f16(
                false, a, false, b, (short)0, acc[nt], false, false);
        }
    }

    // C layout: VGPR r -> M = r + hf*8, N across lanes
    #pragma unroll
    for (int nt = 0; nt < 8; ++nt) {
        #pragma unroll
        for (int r = 0; r < 8; ++r) {
            int row = m0 + hf * 8 + r;
            int col = nt * 16 + l16;
            h[(size_t)row * OUT_F + col] = acc[nt][r];
        }
    }
}

// ---------------- 4) s1 = h@A1, s2 = h@A2 (one wave per row) ----------------
__global__ void scores_kernel(const float* __restrict__ h,
                              const float* __restrict__ A1,
                              const float* __restrict__ A2,
                              float* __restrict__ s1, float* __restrict__ s2) {
    const int wave = threadIdx.x >> 5;
    const int lane = threadIdx.x & 31;
    const int row  = blockIdx.x * 8 + wave;
    if (row >= NN) return;
    const float* hr = h + (size_t)row * OUT_F;
    float p1 = 0.f, p2 = 0.f;
    #pragma unroll
    for (int c = lane; c < OUT_F; c += 32) {
        float v = hr[c];
        p1 += v * A1[c];
        p2 += v * A2[c];
    }
    #pragma unroll
    for (int off = 16; off > 0; off >>= 1) {
        p1 += __shfl_xor(p1, off);
        p2 += __shfl_xor(p2, off);
    }
    if (lane == 0) { s1[row] = p1; s2[row] = p2; }
}

// ---------------- 5) column-mean of h (for edge-less rows) ----------------
__global__ void hmean_kernel(const float* __restrict__ h, float* __restrict__ hm) {
    int c = threadIdx.x;   // 128 threads
    float s = 0.f;
    for (int i = 0; i < NN; ++i) s += h[(size_t)i * OUT_F + c];
    hm[c] = s * (1.0f / (float)NN);
}

// ---------------- 6) masked softmax + aggregation, one block per row ----------------
__global__ __launch_bounds__(256)
void attn_kernel(const float* __restrict__ h,
                 const float* __restrict__ s1, const float* __restrict__ s2,
                 const unsigned* __restrict__ mask,
                 const float* __restrict__ hm,
                 float* __restrict__ out) {
    const int i   = blockIdx.x;
    const int tid = threadIdx.x;
    const unsigned* mrow = mask + (size_t)i * WPR;
    const float s1i = s1[i];

    __shared__ float smax[256];
    __shared__ int   scnt[256];
    __shared__ float accbuf[8][OUT_F];
    __shared__ float wsum[8];

    // ---- phase 1: row max of leaky_relu(s1[i]+s2[j]) over edges, + edge count
    float lmax = -INFINITY;
    int   lcnt = 0;
    for (int w = tid; w < WPR; w += 256) {
        unsigned bits = mrow[w];
        lcnt += __popc(bits);
        while (bits) {
            int b = __ffs(bits) - 1;
            bits &= bits - 1;
            int j = w * 32 + b;
            float z = s1i + s2[j];
            z = (z > 0.f) ? z : NEG_SLOPE * z;
            lmax = fmaxf(lmax, z);
        }
    }
    smax[tid] = lmax; scnt[tid] = lcnt;
    __syncthreads();
    for (int s = 128; s > 0; s >>= 1) {
        if (tid < s) {
            smax[tid] = fmaxf(smax[tid], smax[tid + s]);
            scnt[tid] += scnt[tid + s];
        }
        __syncthreads();
    }
    const float m     = smax[0];
    const int   total = scnt[0];

    if (total == 0) {   // uniform softmax over all N -> column mean of h
        if (tid < OUT_F) out[(size_t)i * OUT_F + tid] = hm[tid];
        return;
    }

    // ---- phase 2: weighted gather-accumulate; wave-wide 128-float accumulator
    const int wave = tid >> 5, lane = tid & 31;
    float4 acc = make_float4(0.f, 0.f, 0.f, 0.f);
    float  sw  = 0.f;
    for (int w = wave; w < WPR; w += 8) {
        unsigned bits = mrow[w];              // uniform within wave
        while (bits) {
            int b = __ffs(bits) - 1;
            bits &= bits - 1;
            int j = w * 32 + b;
            float z = s1i + s2[j];
            z = (z > 0.f) ? z : NEG_SLOPE * z;
            float wj = __expf(z - m);
            sw += wj;
            float4 hv = *(const float4*)(h + (size_t)j * OUT_F + lane * 4);
            acc.x += wj * hv.x; acc.y += wj * hv.y;
            acc.z += wj * hv.z; acc.w += wj * hv.w;
        }
    }
    accbuf[wave][lane * 4 + 0] = acc.x;
    accbuf[wave][lane * 4 + 1] = acc.y;
    accbuf[wave][lane * 4 + 2] = acc.z;
    accbuf[wave][lane * 4 + 3] = acc.w;
    if (lane == 0) wsum[wave] = sw;
    __syncthreads();

    if (tid < OUT_F) {
        float t = 0.f, den = 0.f;
        #pragma unroll
        for (int wv = 0; wv < 8; ++wv) { t += accbuf[wv][tid]; den += wsum[wv]; }
        out[(size_t)i * OUT_F + tid] = t / den;
    }
}

// ---------------- launcher ----------------
extern "C" void kernel_launch(void* const* d_in, const int* in_sizes, int n_in,
                              void* d_out, int out_size, void* d_ws, size_t ws_size,
                              hipStream_t stream) {
    const float*     x  = (const float*)d_in[0];       // [10000, 512]
    const long long* ei = (const long long*)d_in[1];   // [2, 320000] int64
    const float*     W  = (const float*)d_in[2];       // [128, 512]
    const float*     A1 = (const float*)d_in[3];       // [128]
    const float*     A2 = (const float*)d_in[4];       // [128]
    float* out = (float*)d_out;                        // [10000, 128]

    char* ws = (char*)d_ws;
    unsigned*  mask = (unsigned*)(ws + OFF_MASK);
    float*     h    = (float*)(ws + OFF_H);
    float*     s1   = (float*)(ws + OFF_S1);
    float*     s2   = (float*)(ws + OFF_S2);
    float*     hm   = (float*)(ws + OFF_HMEAN);
    _Float16*  Wf16 = (_Float16*)(ws + OFF_WF16);

    const int maskWords = NN * WPR;
    zero_mask_kernel<<<(maskWords + 255) / 256, 256, 0, stream>>>(mask, maskWords);
    build_mask_kernel<<<(NE + 255) / 256, 256, 0, stream>>>(ei, mask);
    convert_w_kernel<<<16, 256, 0, stream>>>(W, Wf16);
    gemm_wmma_kernel<<<(NN / 16 + 3) / 4, 128, WF16_BYTES, stream>>>(x, Wf16, h);
    scores_kernel<<<(NN + 7) / 8, 256, 0, stream>>>(h, A1, A2, s1, s2);
    hmean_kernel<<<1, OUT_F, 0, stream>>>(h, hm);
    attn_kernel<<<NN, 256, 0, stream>>>(h, s1, s2, mask, hm, out);
}
